// Fake_FPN_with_BiFormer_74586402062403
// MI455X (gfx1250) — matrix operand order repaired and checked by reading, not compiled, not run on hardware
//
#include <hip/hip_runtime.h>
#include <math.h>

// ---------------------------------------------------------------------------
// Types for CDNA5 WMMA (wave32, v_wmma_f32_16x16x32_bf16)
// ---------------------------------------------------------------------------
typedef __attribute__((ext_vector_type(16))) __bf16 v16bf;
typedef __attribute__((ext_vector_type(8)))  __bf16 v8bf;
typedef __attribute__((ext_vector_type(8)))  float  v8f;

__device__ __forceinline__ v8f wmma_bf16(v16bf a, v16bf b, v8f c) {
  return __builtin_amdgcn_wmma_f32_16x16x32_bf16(false, a, false, b,
                                                 (short)0, c, false, false);
}

// fp32 -> bf16 hi/lo split (2-term, ~2^-16 relative residual)
__device__ __forceinline__ void splitf(float x, __bf16& h, __bf16& l) {
  h = (__bf16)x;
  l = (__bf16)(x - (float)h);
}

// A-operand per-lane fragment: K chunks {0..7} and {16..23} (two b128 loads)
__device__ __forceinline__ v16bf ld16_a(const __bf16* p) {
  v8bf x = *(const v8bf*)p;
  v8bf y = *(const v8bf*)(p + 16);
  return __builtin_shufflevector(x, y, 0, 1, 2, 3, 4, 5, 6, 7,
                                 8, 9, 10, 11, 12, 13, 14, 15);
}
// B-operand per-lane fragment: K chunks {0..7} and {16..23} (two b128 loads)
__device__ __forceinline__ v16bf ld16_b(const __bf16* p) {
  v8bf x = *(const v8bf*)p;
  v8bf y = *(const v8bf*)(p + 16);
  return __builtin_shufflevector(x, y, 0, 1, 2, 3, 4, 5, 6, 7,
                                 8, 9, 10, 11, 12, 13, 14, 15);
}

// ---------------------------------------------------------------------------
// Pack srcs [b,C,h,w] -> feasF fp32 + feasH/feasL bf16, layout [b, s, c]
// ---------------------------------------------------------------------------
__global__ void pack_feas(const float* __restrict__ s0, const float* __restrict__ s1,
                          const float* __restrict__ s2, const float* __restrict__ s3,
                          float* __restrict__ feasF,
                          __bf16* __restrict__ feasH, __bf16* __restrict__ feasL) {
  size_t tid = (size_t)blockIdx.x * 256 + threadIdx.x; // 2*12240*256 total
  int c = (int)(tid & 255);
  size_t r = tid >> 8;
  int s = (int)(r % 12240);
  int b = (int)(r / 12240);
  const float* sp; int hw, p;
  if (s < 9216)        { sp = s0; hw = 9216; p = s; }
  else if (s < 11520)  { sp = s1; hw = 2304; p = s - 9216; }
  else if (s < 12096)  { sp = s2; hw = 576;  p = s - 11520; }
  else                 { sp = s3; hw = 144;  p = s - 12096; }
  float v = sp[((size_t)b * 256 + c) * hw + p];
  feasF[tid] = v;
  splitf(v, feasH[tid], feasL[tid]);
}

// Generic weight pre-split: fp32 [n] -> bf16 hi/lo
__global__ void pack_w(const float* __restrict__ w, __bf16* __restrict__ hi,
                       __bf16* __restrict__ lo, int n) {
  int tid = blockIdx.x * 256 + threadIdx.x;
  if (tid >= n) return;
  splitf(w[tid], hi[tid], lo[tid]);
}

// ---------------------------------------------------------------------------
// Fused sim GEMM + running top-4.  One wave per 16 queries of one batch.
// A panel register-resident; inner loop = 4x b128 loads + 3 WMMAs per k-step.
// ---------------------------------------------------------------------------
__global__ void __launch_bounds__(32)
sim_topk(const __bf16* __restrict__ feasH, const __bf16* __restrict__ feasL,
         int hw, int qoff, int* __restrict__ topk) {
  const int S = 12240;
  int lane = threadIdx.x;
  int hf = lane >> 4, l15 = lane & 15;
  int b = blockIdx.y;
  int m0 = blockIdx.x * 16;

  size_t arow = ((size_t)(b * S + qoff + m0 + l15)) * 256 + hf * 8;
  v16bf ah[8], al[8];
#pragma unroll
  for (int ks = 0; ks < 8; ++ks) {
    ah[ks] = ld16_a(feasH + arow + ks * 32);
    al[ks] = ld16_a(feasL + arow + ks * 32);
  }

  size_t bbase = (size_t)b * S * 256 + hf * 8;

  float tv[4] = {-3.402823466e38f, -3.402823466e38f, -3.402823466e38f, -3.402823466e38f};
  int   ti[4] = {0, 0, 0, 0};
  __shared__ float sm[256];

  for (int st = 0; st < S / 16; ++st) {
    int s0 = st * 16;
    size_t bcol = bbase + (size_t)(s0 + l15) * 256;
    v8f acc = {};
#pragma unroll
    for (int ks = 0; ks < 8; ++ks) {
      v16bf bh = ld16_b(feasH + bcol + ks * 32);
      v16bf bl = ld16_b(feasL + bcol + ks * 32);
      acc = wmma_bf16(ah[ks], bh, acc);
      acc = wmma_bf16(ah[ks], bl, acc);
      acc = wmma_bf16(al[ks], bh, acc);
    }
    __syncthreads();
#pragma unroll
    for (int i = 0; i < 8; ++i) sm[(hf * 8 + i) * 16 + l15] = acc[i];
    __syncthreads();
    if (lane < 16) {
#pragma unroll 4
      for (int j = 0; j < 16; ++j) {
        float v = sm[lane * 16 + j];
        if (v > tv[3]) {                 // strict > keeps earlier index on ties
          int si = s0 + j;
          int pos = 3;
          while (pos > 0 && v > tv[pos - 1]) {
            tv[pos] = tv[pos - 1]; ti[pos] = ti[pos - 1]; --pos;
          }
          tv[pos] = v; ti[pos] = si;
        }
      }
    }
  }
  if (lane < 16) {
    size_t n = (size_t)b * hw + m0 + lane;
#pragma unroll
    for (int j = 0; j < 4; ++j) topk[n * 4 + j] = ti[j];
  }
}

// ---------------------------------------------------------------------------
// Generic out = A[M,K] @ W[N,K]^T + bias; A,W pre-split bf16 hi/lo.
// Optional fp32 out, optional packed out (for GEMM->GEMM chains), opt. relu.
// ---------------------------------------------------------------------------
__global__ void __launch_bounds__(32)
gemm_wmma(const __bf16* __restrict__ Ah, const __bf16* __restrict__ Al, int lda,
          const __bf16* __restrict__ Wh, const __bf16* __restrict__ Wl, int ldw,
          const float* __restrict__ bias,
          float* __restrict__ out, __bf16* __restrict__ outH, __bf16* __restrict__ outL,
          int ldo, int ksteps, int relu) {
  int lane = threadIdx.x;
  int hf = lane >> 4, l15 = lane & 15;
  size_t m0 = (size_t)blockIdx.x * 16;
  int n0 = blockIdx.y * 16;
  size_t arow = (m0 + l15) * (size_t)lda + hf * 8;
  size_t brow = (size_t)(n0 + l15) * ldw + hf * 8;
  v8f acc = {};
  for (int ks = 0; ks < ksteps; ++ks) {
    v16bf ahv = ld16_a(Ah + arow + ks * 32);
    v16bf alv = ld16_a(Al + arow + ks * 32);
    v16bf bhv = ld16_b(Wh + brow + ks * 32);
    v16bf blv = ld16_b(Wl + brow + ks * 32);
    acc = wmma_bf16(ahv, bhv, acc);
    acc = wmma_bf16(ahv, blv, acc);
    acc = wmma_bf16(alv, bhv, acc);
  }
  int n = n0 + l15;
  float bb = bias ? bias[n] : 0.0f;
#pragma unroll
  for (int i = 0; i < 8; ++i) {
    float v = acc[i] + bb;
    if (relu) v = fmaxf(v, 0.0f);
    size_t o = (m0 + hf * 8 + i) * (size_t)ldo + n;
    if (out) out[o] = v;
    if (outH) splitf(v, outH[o], outL[o]);
  }
}

// ---------------------------------------------------------------------------
// Query slice -> X fp32 + Xh/Xl;  kv gather (batch-0 rows, faithful to the
// reference's un-offset index_select) -> packed KV only.
// ---------------------------------------------------------------------------
__global__ void copy_x(const float* __restrict__ feasF, float* __restrict__ X,
                       __bf16* __restrict__ Xh, __bf16* __restrict__ Xl,
                       int hw, int qoff) {
  size_t tid = (size_t)blockIdx.x * 256 + threadIdx.x; // N*256 total
  int c = (int)(tid & 255);
  size_t n = tid >> 8;
  int b = (int)(n / hw), p = (int)(n % hw);
  float v = feasF[((size_t)(b * 12240 + qoff + p)) * 256 + c];
  X[tid] = v;
  splitf(v, Xh[tid], Xl[tid]);
}

__global__ void gather_kv(const __bf16* __restrict__ feasH, const __bf16* __restrict__ feasL,
                          const int* __restrict__ topk,
                          __bf16* __restrict__ KVh, __bf16* __restrict__ KVl) {
  size_t tid = (size_t)blockIdx.x * 256 + threadIdx.x; // 4N*256 total
  int c = (int)(tid & 255);
  size_t r = tid >> 8;                 // r = n*4 + k
  size_t si = (size_t)topk[r] * 256 + c;  // batch-0 rows
  KVh[tid] = feasH[si];
  KVl[tid] = feasL[si];
}

// ---------------------------------------------------------------------------
// K=4, 8-head cross attention.  One thread per (query, head). Packed output.
// ---------------------------------------------------------------------------
__global__ void attn_kernel(const float* __restrict__ Q, const float* __restrict__ KP,
                            const float* __restrict__ VP,
                            __bf16* __restrict__ CAh, __bf16* __restrict__ CAl, int N) {
  int t = blockIdx.x * blockDim.x + threadIdx.x;
  if (t >= N * 8) return;
  int n = t >> 3, h = t & 7;
  const float scale = 0.1767766952966369f; // 1/sqrt(32)
  float q[32];
  const float* qp = Q + (size_t)n * 256 + h * 32;
#pragma unroll
  for (int d = 0; d < 32; ++d) q[d] = qp[d];
  float lg[4];
#pragma unroll
  for (int k = 0; k < 4; ++k) {
    const float* kp = KP + ((size_t)n * 4 + k) * 256 + h * 32;
    float s = 0.0f;
#pragma unroll
    for (int d = 0; d < 32; ++d) s += q[d] * kp[d];
    lg[k] = s * scale;
  }
  float m = fmaxf(fmaxf(lg[0], lg[1]), fmaxf(lg[2], lg[3]));
  float a[4], ssum = 0.0f;
#pragma unroll
  for (int k = 0; k < 4; ++k) { a[k] = expf(lg[k] - m); ssum += a[k]; }
  float inv = 1.0f / ssum;
#pragma unroll
  for (int k = 0; k < 4; ++k) a[k] *= inv;
  size_t ob = (size_t)n * 256 + h * 32;
#pragma unroll
  for (int d = 0; d < 32; ++d) {
    float o = 0.0f;
#pragma unroll
    for (int k = 0; k < 4; ++k) o += a[k] * VP[((size_t)n * 4 + k) * 256 + h * 32 + d];
    splitf(o, CAh[ob + d], CAl[ob + d]);
  }
}

// ---------------------------------------------------------------------------
// Fused residual + LayerNorm, one wave per row of 256.
// ---------------------------------------------------------------------------
__device__ __forceinline__ float wave_sum(float s) {
#pragma unroll
  for (int m = 16; m >= 1; m >>= 1) s += __shfl_xor(s, m, 32);
  return s;
}

// X <- LN(X + D); also emits packed Xh/Xl for downstream GEMMs.
__global__ void __launch_bounds__(32)
resln(float* __restrict__ X, const float* __restrict__ D,
      const float* __restrict__ g, const float* __restrict__ b,
      __bf16* __restrict__ Xh, __bf16* __restrict__ Xl) {
  int row = blockIdx.x, lane = threadIdx.x;
  size_t base = (size_t)row * 256;
  float v[8];
  float s = 0.0f;
#pragma unroll
  for (int i = 0; i < 8; ++i) {
    int c = i * 32 + lane;
    v[i] = X[base + c] + D[base + c];
    s += v[i];
  }
  float mean = wave_sum(s) * (1.0f / 256.0f);
  float q = 0.0f;
#pragma unroll
  for (int i = 0; i < 8; ++i) { float d = v[i] - mean; q += d * d; }
  float var = wave_sum(q) * (1.0f / 256.0f);
  float rs = 1.0f / sqrtf(var + 1e-5f);
#pragma unroll
  for (int i = 0; i < 8; ++i) {
    int c = i * 32 + lane;
    float y = (v[i] - mean) * rs * g[c] + b[c];
    X[base + c] = y;
    splitf(y, Xh[base + c], Xl[base + c]);
  }
}

// Final: out[b,c,p] = LN(X + D)[n,c] + src[b,c,p]   (n = b*hw + p)
__global__ void __launch_bounds__(32)
resln_out(const float* __restrict__ X, const float* __restrict__ D,
          const float* __restrict__ g, const float* __restrict__ b,
          const float* __restrict__ src, float* __restrict__ out, int hw) {
  int row = blockIdx.x, lane = threadIdx.x;
  int bb = row / hw, p = row % hw;
  size_t base = (size_t)row * 256;
  float v[8];
  float s = 0.0f;
#pragma unroll
  for (int i = 0; i < 8; ++i) {
    int c = i * 32 + lane;
    v[i] = X[base + c] + D[base + c];
    s += v[i];
  }
  float mean = wave_sum(s) * (1.0f / 256.0f);
  float q = 0.0f;
#pragma unroll
  for (int i = 0; i < 8; ++i) { float d = v[i] - mean; q += d * d; }
  float var = wave_sum(q) * (1.0f / 256.0f);
  float rs = 1.0f / sqrtf(var + 1e-5f);
#pragma unroll
  for (int i = 0; i < 8; ++i) {
    int c = i * 32 + lane;
    float y = (v[i] - mean) * rs * g[c] + b[c];
    size_t so = ((size_t)bb * 256 + c) * hw + p;
    out[so] = y + src[so];
  }
}

// ---------------------------------------------------------------------------
// Orchestration
// ---------------------------------------------------------------------------
extern "C" void kernel_launch(void* const* d_in, const int* in_sizes, int n_in,
                              void* d_out, int out_size, void* d_ws, size_t ws_size,
                              hipStream_t stream) {
  (void)in_sizes; (void)n_in; (void)out_size; (void)ws_size;
  const float* src[4] = {(const float*)d_in[0], (const float*)d_in[1],
                         (const float*)d_in[2], (const float*)d_in[3]};
  const float* sa_in_w  = (const float*)d_in[4];
  const float* sa_in_b  = (const float*)d_in[5];
  const float* sa_out_w = (const float*)d_in[6];
  const float* sa_out_b = (const float*)d_in[7];
  const float* ca_in_w  = (const float*)d_in[8];
  const float* ca_in_b  = (const float*)d_in[9];
  const float* ca_out_w = (const float*)d_in[10];
  const float* ca_out_b = (const float*)d_in[11];
  const float* lin1_w   = (const float*)d_in[12];
  const float* lin1_b   = (const float*)d_in[13];
  const float* lin2_w   = (const float*)d_in[14];
  const float* lin2_b   = (const float*)d_in[15];
  const float* ln1_g = (const float*)d_in[16];
  const float* ln1_b = (const float*)d_in[17];
  const float* ln2_g = (const float*)d_in[18];
  const float* ln2_b = (const float*)d_in[19];
  const float* ln3_g = (const float*)d_in[20];
  const float* ln3_b = (const float*)d_in[21];

  // Workspace arena (float-unit offsets; bf16 arrays use n/2 floats).
  float* ws = (float*)d_ws;
  float*  feasF = ws;                               // 6,266,880 f32
  __bf16* feasH = (__bf16*)(feasF + 6266880);       // 6,266,880 bf16
  __bf16* feasL = (__bf16*)((float*)feasH + 3133440);
  float*  Xf    = (float*)feasL + 3133440;          // 1,179,648 f32
  __bf16* Xh    = (__bf16*)(Xf + 1179648);          // 1,179,648 bf16
  __bf16* Xl    = (__bf16*)((float*)Xh + 589824);
  float*  T1f   = (float*)Xl + 589824;              // 1,179,648 f32
  __bf16* T1h   = (__bf16*)(T1f + 1179648);         // 4,718,592 bf16 (N*1024)
  __bf16* T1l   = (__bf16*)((float*)T1h + 2359296);
  float*  T2f   = (float*)T1l + 2359296;            // 1,179,648 f32
  __bf16* T2h   = (__bf16*)(T2f + 1179648);         // 1,179,648 bf16
  __bf16* T2l   = (__bf16*)((float*)T2h + 589824);
  __bf16* KVh   = (__bf16*)((float*)T2l + 589824);  // 4,718,592 bf16
  __bf16* KVl   = (__bf16*)((float*)KVh + 2359296);
  float*  KPf   = (float*)KVl + 2359296;            // 4,718,592 f32
  float*  VPf   = KPf + 4718592;                    // 4,718,592 f32
  __bf16* WH    = (__bf16*)(VPf + 4718592);         // 3,145,728 bf16
  __bf16* WL    = (__bf16*)((float*)WH + 1572864);
  int*    topk  = (int*)((float*)WL + 1572864);     // 18,432 ints

  // Packed-weight arena offsets (bf16 elements)
  const size_t O_SAW = 0;        // sa_in_w  3*768*256
  const size_t O_SOW = 589824;   // sa_out_w 3*256*256
  const size_t O_CAW = 786432;   // ca_in_w  3*768*256
  const size_t O_COW = 1376256;  // ca_out_w 3*256*256
  const size_t O_L1W = 1572864;  // lin1_w   3*1024*256
  const size_t O_L2W = 2359296;  // lin2_w   3*256*1024

  pack_feas<<<24480, 256, 0, stream>>>(src[0], src[1], src[2], src[3],
                                       feasF, feasH, feasL);
  pack_w<<<2304, 256, 0, stream>>>(sa_in_w,  WH + O_SAW, WL + O_SAW, 589824);
  pack_w<<<768,  256, 0, stream>>>(sa_out_w, WH + O_SOW, WL + O_SOW, 196608);
  pack_w<<<2304, 256, 0, stream>>>(ca_in_w,  WH + O_CAW, WL + O_CAW, 589824);
  pack_w<<<768,  256, 0, stream>>>(ca_out_w, WH + O_COW, WL + O_COW, 196608);
  pack_w<<<3072, 256, 0, stream>>>(lin1_w,   WH + O_L1W, WL + O_L1W, 786432);
  pack_w<<<3072, 256, 0, stream>>>(lin2_w,   WH + O_L2W, WL + O_L2W, 786432);

  const int HW[3]      = {144, 576, 2304};
  const int OFF[3]     = {12096, 11520, 9216};
  const size_t OOFF[3] = {1474560, 1179648, 0};

  for (int l = 0; l < 3; ++l) {
    int hw = HW[l], off = OFF[l], N = 2 * hw;
    const float* srcq = src[3 - l];

    size_t wv_sa = O_SAW + (size_t)l * 768 * 256 + 512 * 256;
    size_t saow  = O_SOW + (size_t)l * 65536;
    size_t wq    = O_CAW + (size_t)l * 768 * 256;
    size_t wk    = wq + 65536;
    size_t wvv   = wk + 65536;
    size_t caow  = O_COW + (size_t)l * 65536;
    size_t l1w   = O_L1W + (size_t)l * 1024 * 256;
    size_t l2w   = O_L2W + (size_t)l * 256 * 1024;

    const float* bv_sa = sa_in_b + l * 768 + 512;
    const float* saob  = sa_out_b + l * 256;
    const float* bq    = ca_in_b + l * 768;
    const float* bk    = bq + 256;
    const float* bvv   = bk + 256;
    const float* caob  = ca_out_b + l * 256;
    const float* l1b   = lin1_b + l * 1024;
    const float* l2b   = lin2_b + l * 256;

    // sim + top-4 (register-resident A panel, pure vmem+wmma inner loop)
    sim_topk<<<dim3(hw / 16, 2), 32, 0, stream>>>(feasH, feasL, hw, off, topk);
    copy_x<<<N, 256, 0, stream>>>(feasF, Xf, Xh, Xl, hw, off);
    gather_kv<<<4 * N, 256, 0, stream>>>(feasH, feasL, topk, KVh, KVl);

    // self-attn path (seq-len 1 => out_proj(v_proj(x)))
    gemm_wmma<<<dim3(N / 16, 16), 32, 0, stream>>>(Xh, Xl, 256, WH + wv_sa, WL + wv_sa, 256,
        bv_sa, nullptr, T2h, T2l, 256, 8, 0);
    gemm_wmma<<<dim3(N / 16, 16), 32, 0, stream>>>(T2h, T2l, 256, WH + saow, WL + saow, 256,
        saob, T1f, nullptr, nullptr, 256, 8, 0);
    resln<<<N, 32, 0, stream>>>(Xf, T1f, ln1_g + l * 256, ln1_b + l * 256, Xh, Xl);

    // cross-attn projections
    gemm_wmma<<<dim3(N / 16, 16), 32, 0, stream>>>(Xh, Xl, 256, WH + wq, WL + wq, 256,
        bq, T2f, nullptr, nullptr, 256, 8, 0);
    gemm_wmma<<<dim3(4 * N / 16, 16), 32, 0, stream>>>(KVh, KVl, 256, WH + wk, WL + wk, 256,
        bk, KPf, nullptr, nullptr, 256, 8, 0);
    gemm_wmma<<<dim3(4 * N / 16, 16), 32, 0, stream>>>(KVh, KVl, 256, WH + wvv, WL + wvv, 256,
        bvv, VPf, nullptr, nullptr, 256, 8, 0);
    attn_kernel<<<(N * 8 + 255) / 256, 256, 0, stream>>>(T2f, KPf, VPf, T1h, T1l, N);
    gemm_wmma<<<dim3(N / 16, 16), 32, 0, stream>>>(T1h, T1l, 256, WH + caow, WL + caow, 256,
        caob, T2f, nullptr, nullptr, 256, 8, 0);
    resln<<<N, 32, 0, stream>>>(Xf, T2f, ln2_g + l * 256, ln2_b + l * 256, Xh, Xl);

    // MLP
    gemm_wmma<<<dim3(N / 16, 64), 32, 0, stream>>>(Xh, Xl, 256, WH + l1w, WL + l1w, 256,
        l1b, nullptr, T1h, T1l, 1024, 8, 1);
    gemm_wmma<<<dim3(N / 16, 16), 32, 0, stream>>>(T1h, T1l, 1024, WH + l2w, WL + l2w, 1024,
        l2b, T2f, nullptr, nullptr, 256, 32, 0);

    // final LN + residual + transpose-store (+ src residual)
    resln_out<<<N, 32, 0, stream>>>(Xf, T2f, ln3_g + l * 256, ln3_b + l * 256,
                                    srcq, (float*)d_out + OOFF[l], hw);
  }
}
